// MultiHeadAttention_44014824849982
// MI455X (gfx1250) — compile-verified, matrix operand order checked by reading, not run on hardware
//
#include <hip/hip_runtime.h>
#include <hip/hip_bf16.h>

typedef __attribute__((ext_vector_type(16))) __bf16 v16bf;
typedef __attribute__((ext_vector_type(8)))  float  v8f;
typedef __attribute__((ext_vector_type(4)))  int    v4i;

#define WMMA_BF16(a, b, c) \
  __builtin_amdgcn_wmma_f32_16x16x32_bf16(false, (a), false, (b), (short)0, (c), false, false)

// --- CDNA5 async global->LDS copy (ASYNCcnt path), with safe fallback -------
#if defined(__has_builtin)
#if __has_builtin(__builtin_amdgcn_global_load_async_to_lds_b128) && \
    __has_builtin(__builtin_amdgcn_s_wait_asynccnt)
#define USE_ASYNC_LDS 1
#endif
#endif
#ifndef USE_ASYNC_LDS
#define USE_ASYNC_LDS 0
#endif

typedef __attribute__((address_space(1))) v4i* gv4i_p;   // global int4*
typedef __attribute__((address_space(3))) v4i* lv4i_p;   // LDS int4*

__device__ __forceinline__ void copy16_g2l(void* lds, const void* g) {
#if USE_ASYNC_LDS
  __builtin_amdgcn_global_load_async_to_lds_b128((gv4i_p)(g), (lv4i_p)(lds), 0, 0);
#else
  *(uint4*)lds = *(const uint4*)g;
#endif
}
__device__ __forceinline__ void async_wait_all() {
#if USE_ASYNC_LDS
  __builtin_amdgcn_s_wait_asynccnt(0);
#endif
}

// Load a 16x32 A/B-fragment (lane = row/col index, 32 contiguous bf16 along K).
// ISA 7.12.2: lanes 0-15 hold K=0..7 in v0..3 and K=16..23 in v4..7;
// lanes 16-31 hold K=8..15 / K=24..31.
__device__ __forceinline__ v16bf frag32(const __bf16* rowPtr, int half) {
  const unsigned int* p = (const unsigned int*)rowPtr;  // pair j = K{2j,2j+1}
  const int kb = half * 4;
  union { v16bf v; unsigned int u[8]; } f;
#pragma unroll
  for (int j = 0; j < 4; ++j) {
    f.u[j]     = p[kb + j];
    f.u[4 + j] = p[8 + kb + j];
  }
  return f.v;
}

// ---------------------------------------------------------------------------
// Kernel 1: per-head projection.  Y[b,h,n,:] = scale * (X[b,n,:] @ W[h,:,:])
// X: [8192][512] f32, W: [H][512][64] f32, Y: [b][h][n][64] bf16
// block = 256 (8 waves), tile = 128 rows x 64 cols, K-step 32.
// ---------------------------------------------------------------------------
__global__ void mha_qkv_proj(const float* __restrict__ X,
                             const float* __restrict__ Wall,
                             __bf16* __restrict__ Y, float scale) {
  __shared__ __bf16 As[128][32];
  __shared__ __bf16 Bs[64][32];   // transposed: [out-col][k]
  const int h = blockIdx.y;
  const int rowBase = blockIdx.x * 128;
  const int tid = threadIdx.x;
  const int wave = tid >> 5, lane = tid & 31;
  const int half = lane >> 4, rsel = lane & 15;
  const float* W = Wall + (size_t)h * 512 * 64;

  v8f acc[4] = {v8f{}, v8f{}, v8f{}, v8f{}};

  for (int kk = 0; kk < 512; kk += 32) {
    // A tile: 128x32 f32 -> bf16 LDS. 1024 float4, 4 per thread.
#pragma unroll
    for (int i = 0; i < 4; ++i) {
      int idx = tid + i * 256;
      int r = idx >> 3, c4 = (idx & 7) * 4;
      float4 f = *(const float4*)(X + (size_t)(rowBase + r) * 512 + kk + c4);
      __bf16* d = &As[r][c4];
      d[0] = (__bf16)f.x; d[1] = (__bf16)f.y; d[2] = (__bf16)f.z; d[3] = (__bf16)f.w;
    }
    // W tile: 32x64 f32 -> transposed bf16 LDS. 512 float4, 2 per thread.
#pragma unroll
    for (int i = 0; i < 2; ++i) {
      int idx = tid + i * 256;
      int kr = idx >> 4, c4 = (idx & 15) * 4;
      float4 f = *(const float4*)(W + (size_t)(kk + kr) * 64 + c4);
      Bs[c4 + 0][kr] = (__bf16)f.x; Bs[c4 + 1][kr] = (__bf16)f.y;
      Bs[c4 + 2][kr] = (__bf16)f.z; Bs[c4 + 3][kr] = (__bf16)f.w;
    }
    __syncthreads();
    v16bf a = frag32(&As[wave * 16 + rsel][0], half);
#pragma unroll
    for (int c = 0; c < 4; ++c) {
      v16bf b = frag32(&Bs[c * 16 + rsel][0], half);
      acc[c] = WMMA_BF16(a, b, acc[c]);
    }
    __syncthreads();
  }
#pragma unroll
  for (int c = 0; c < 4; ++c)
#pragma unroll
    for (int r = 0; r < 8; ++r) {
      int rowG = rowBase + wave * 16 + r + 8 * half;
      int b = rowG >> 11, n = rowG & 2047;
      Y[(((size_t)b * 8 + h) * 2048 + n) * 64 + c * 16 + rsel] =
          (__bf16)(acc[c][r] * scale);
    }
}

// ---------------------------------------------------------------------------
// Kernel 2: flash attention per (b,h).  Q,K,V: [b][h][2048][64] bf16.
// block = 128 (4 waves); each wave owns 16 query rows; key tiles of 64.
// Output Ow: [b][n][h*64 + hs] bf16.
// ---------------------------------------------------------------------------
__global__ void mha_flash_attn(const __bf16* __restrict__ Qp,
                               const __bf16* __restrict__ Kp,
                               const __bf16* __restrict__ Vp,
                               __bf16* __restrict__ Ow) {
  __shared__ __bf16 Kt[64][64];      // [key][hs]  (B-frag for Q@K^T)
  __shared__ __bf16 Vt[64][68];      // [hs][key]  (B-frag for P@V), padded
  __shared__ __bf16 Pl[4][16][64];   // per-wave P transpose buffer

  const int bh = blockIdx.y;                       // b*8 + h
  const size_t base = (size_t)bh * 2048 * 64;
  const int qBase = blockIdx.x * 64;
  const int tid = threadIdx.x;
  const int wave = tid >> 5, lane = tid & 31;
  const int half = lane >> 4, rsel = lane & 15;

  // Q fragments (already scaled by 1/sqrt(HS) at projection time)
  const __bf16* qrow = Qp + base + (size_t)(qBase + wave * 16 + rsel) * 64;
  v16bf qa0 = frag32(qrow, half);
  v16bf qa1 = frag32(qrow + 32, half);

  v8f O[4] = {v8f{}, v8f{}, v8f{}, v8f{}};
  float mrow[8], lrow[8];
#pragma unroll
  for (int r = 0; r < 8; ++r) { mrow[r] = -3.0e38f; lrow[r] = 0.0f; }

  for (int j = 0; j < 2048; j += 64) {
    const __bf16* Kg = Kp + base + (size_t)j * 64;
    const __bf16* Vg = Vp + base + (size_t)j * 64;
    // K tile: 4096 bf16 = 512 x b128, async direct to LDS, 4 per thread.
#pragma unroll
    for (int i = 0; i < 4; ++i) {
      int idx = tid + i * 128;
      int r = idx >> 3, c8 = (idx & 7) * 8;
      copy16_g2l(&Kt[r][c8], Kg + r * 64 + c8);
    }
    // V tile transposed: work item = (row-pair, hs-octet); two b128 loads,
    // eight packed b32 LDS stores {V[m][hs], V[m+1][hs]}.
#pragma unroll
    for (int i = 0; i < 2; ++i) {
      int idx = tid + i * 128;               // 0..255
      int m = (idx >> 3) * 2, hs0 = (idx & 7) * 8;
      union { uint4 q; unsigned short s[8]; } a, b;
      a.q = *(const uint4*)(Vg + m * 64 + hs0);
      b.q = *(const uint4*)(Vg + (m + 1) * 64 + hs0);
#pragma unroll
      for (int e = 0; e < 8; ++e) {
        unsigned int w = (unsigned int)a.s[e] | ((unsigned int)b.s[e] << 16);
        *(unsigned int*)&Vt[hs0 + e][m] = w;
      }
    }
    async_wait_all();
    __syncthreads();

    if (j + 64 < 2048) {   // global_prefetch_b8 for next tiles
      __builtin_prefetch(Kg + 64 * 64, 0, 1);
      __builtin_prefetch(Vg + 64 * 64, 0, 1);
    }

    // S = Q @ K^T  (16 x 64 per wave, 4 subtiles, K=64 chained as 2 wmma)
    v8f S[4];
#pragma unroll
    for (int c = 0; c < 4; ++c) {
      v16bf kb0 = frag32(&Kt[c * 16 + rsel][0], half);
      v16bf kb1 = frag32(&Kt[c * 16 + rsel][32], half);
      v8f s = {};
      s = WMMA_BF16(qa0, kb0, s);
      s = WMMA_BF16(qa1, kb1, s);
      S[c] = s;
    }

    // Online softmax: row r+8*half lives across 16 lanes (xor 1,2,4,8 stays
    // inside the half-group under wave32).
#pragma unroll
    for (int r = 0; r < 8; ++r) {
      float mx = fmaxf(fmaxf(S[0][r], S[1][r]), fmaxf(S[2][r], S[3][r]));
#pragma unroll
      for (int m = 1; m < 16; m <<= 1) mx = fmaxf(mx, __shfl_xor(mx, m, 32));
      float mn = fmaxf(mrow[r], mx);
      float sf = __expf(mrow[r] - mn);
      mrow[r] = mn;
      float rs = 0.0f;
#pragma unroll
      for (int c = 0; c < 4; ++c) {
        float p = __expf(S[c][r] - mn);
        S[c][r] = p;
        rs += p;
      }
#pragma unroll
      for (int m = 1; m < 16; m <<= 1) rs += __shfl_xor(rs, m, 32);
      lrow[r] = lrow[r] * sf + rs;
#pragma unroll
      for (int c = 0; c < 4; ++c) O[c][r] *= sf;
    }

    // C-layout -> A-layout for P via per-wave LDS (DS ops in-order per wave).
#pragma unroll
    for (int c = 0; c < 4; ++c)
#pragma unroll
      for (int r = 0; r < 8; ++r)
        Pl[wave][r + 8 * half][c * 16 + rsel] = (__bf16)S[c][r];
    v16bf pa0 = frag32(&Pl[wave][rsel][0], half);
    v16bf pa1 = frag32(&Pl[wave][rsel][32], half);

    // O += P @ V  (K = 64 keys, 2 chained wmma per hs subtile)
#pragma unroll
    for (int c = 0; c < 4; ++c) {
      v16bf vb0 = frag32(&Vt[c * 16 + rsel][0], half);
      v16bf vb1 = frag32(&Vt[c * 16 + rsel][32], half);
      O[c] = WMMA_BF16(pa0, vb0, O[c]);
      O[c] = WMMA_BF16(pa1, vb1, O[c]);
    }
    __syncthreads();   // protect Kt/Vt before next tile's loads
  }

  // Normalize and scatter to [b][n][h*64+hs] bf16.
  const int b = bh >> 3, h = bh & 7;
#pragma unroll
  for (int r = 0; r < 8; ++r) {
    float inv = 1.0f / lrow[r];
    int n = qBase + wave * 16 + r + 8 * half;
#pragma unroll
    for (int c = 0; c < 4; ++c)
      Ow[((size_t)b * 2048 + n) * 512 + h * 64 + c * 16 + rsel] =
          (__bf16)(O[c][r] * inv);
  }
}

// ---------------------------------------------------------------------------
// Kernel 3: output projection. out[r,:] = Ow[r,:] @ Wp + bias
// Ow: [8192][512] bf16, Wp: [512][512] f32, out f32.
// block = 256 (8 waves), tile 128 x 64, K-step 32.
// ---------------------------------------------------------------------------
__global__ void mha_out_proj(const __bf16* __restrict__ A,
                             const float* __restrict__ W,
                             const float* __restrict__ bias,
                             float* __restrict__ out) {
  __shared__ __bf16 As[128][32];
  __shared__ __bf16 Bs[64][32];
  const int tid = threadIdx.x;
  const int wave = tid >> 5, lane = tid & 31;
  const int half = lane >> 4, rsel = lane & 15;
  const int rowBase = blockIdx.x * 128;
  const int colBase = blockIdx.y * 64;

  v8f acc[4] = {v8f{}, v8f{}, v8f{}, v8f{}};

  for (int kk = 0; kk < 512; kk += 32) {
    // A tile: 128x32 bf16 = 512 x b128, async direct to LDS, 2 per thread.
#pragma unroll
    for (int i = 0; i < 2; ++i) {
      int idx = tid + i * 256;
      int r = idx >> 2, c8 = (idx & 3) * 8;
      copy16_g2l(&As[r][c8], A + (size_t)(rowBase + r) * 512 + kk + c8);
    }
    // W tile: 32x64 f32 -> transposed bf16 LDS.
#pragma unroll
    for (int i = 0; i < 2; ++i) {
      int idx = tid + i * 256;
      int kr = idx >> 4, c4 = (idx & 15) * 4;
      float4 f = *(const float4*)(W + (size_t)(kk + kr) * 512 + colBase + c4);
      Bs[c4 + 0][kr] = (__bf16)f.x; Bs[c4 + 1][kr] = (__bf16)f.y;
      Bs[c4 + 2][kr] = (__bf16)f.z; Bs[c4 + 3][kr] = (__bf16)f.w;
    }
    async_wait_all();
    __syncthreads();
    v16bf a = frag32(&As[wave * 16 + rsel][0], half);
#pragma unroll
    for (int c = 0; c < 4; ++c) {
      v16bf b = frag32(&Bs[c * 16 + rsel][0], half);
      acc[c] = WMMA_BF16(a, b, acc[c]);
    }
    __syncthreads();
  }
#pragma unroll
  for (int c = 0; c < 4; ++c)
#pragma unroll
    for (int r = 0; r < 8; ++r) {
      int rowG = rowBase + wave * 16 + r + 8 * half;
      int col = colBase + c * 16 + rsel;
      out[(size_t)rowG * 512 + col] = acc[c][r] + bias[col];
    }
}

// ---------------------------------------------------------------------------
extern "C" void kernel_launch(void* const* d_in, const int* in_sizes, int n_in,
                              void* d_out, int out_size, void* d_ws, size_t ws_size,
                              hipStream_t stream) {
  (void)in_sizes; (void)n_in; (void)out_size; (void)ws_size;
  const float* query = (const float*)d_in[0];
  const float* key   = (const float*)d_in[1];
  const float* value = (const float*)d_in[2];
  const float* qk_w  = (const float*)d_in[3];
  const float* kk_w  = (const float*)d_in[4];
  const float* vk_w  = (const float*)d_in[5];
  const float* pk_w  = (const float*)d_in[6];
  const float* pbias = (const float*)d_in[7];
  float* out = (float*)d_out;

  // Workspace: Qp/Kp/Vp [b][h][2048][64] bf16, Ow [b][2048][512] bf16 = 32 MB
  const size_t nProj = (size_t)4 * 8 * 2048 * 64;
  __bf16* Qp = (__bf16*)d_ws;
  __bf16* Kp = Qp + nProj;
  __bf16* Vp = Kp + nProj;
  __bf16* Ow = Vp + nProj;

  dim3 gp(64, 8);   // 8192/128 row tiles x 8 heads
  mha_qkv_proj<<<gp, 256, 0, stream>>>(query, qk_w, Qp, 0.125f);  // 1/sqrt(64)
  mha_qkv_proj<<<gp, 256, 0, stream>>>(key,   kk_w, Kp, 1.0f);
  mha_qkv_proj<<<gp, 256, 0, stream>>>(value, vk_w, Vp, 1.0f);

  dim3 ga(32, 32);  // 2048/64 query tiles x (B*H)
  mha_flash_attn<<<ga, 128, 0, stream>>>(Qp, Kp, Vp, Ow);

  dim3 go(64, 8);   // 8192/128 row tiles x 512/64 col tiles
  mha_out_proj<<<go, 256, 0, stream>>>(Ow, pk_w, pbias, out);
}
